// MimiResidualVectorQuantizer_18983755448881
// MI455X (gfx1250) — matrix operand order repaired and checked by reading, not compiled, not run on hardware
//
#include <hip/hip_runtime.h>

#define TT   32768
#define DIN  512
#define DCB  256
#define KC   2048
#define NQ   8

typedef __bf16 bf16_t;
typedef bf16_t v16bf __attribute__((ext_vector_type(16)));
typedef bf16_t v8bf  __attribute__((ext_vector_type(8)));
typedef float  v8f   __attribute__((ext_vector_type(8)));
typedef unsigned int u32x4 __attribute__((ext_vector_type(4)));
typedef int          i32x4 __attribute__((ext_vector_type(4)));
typedef int          i32x8 __attribute__((ext_vector_type(8)));

union V16  { v16bf v; v8bf h[2]; };
union V16U { v16bf v; unsigned short s[16]; };

// f32 -> bf16 bits, round-to-nearest-even
__device__ __forceinline__ unsigned short f2bf(float f) {
  unsigned u = __float_as_uint(f);
  return (unsigned short)((u + 0x7FFFu + ((u >> 16) & 1u)) >> 16);
}

// A fragment (16x32 bf16, M x K): lane l -> row m0+(l&15); element e -> k = k0 + (e>>3)*16 + (l>>4)*8 + (e&7)
__device__ __forceinline__ v16bf load_a_frag(const unsigned short* base, int ld, int m0, int k0, int lane) {
  int m  = m0 + (lane & 15);
  int hi = lane >> 4;
  const unsigned short* rp = base + m * ld + k0 + hi * 8;
  V16 u;
  u.h[0] = *reinterpret_cast<const v8bf*>(rp);
  u.h[1] = *reinterpret_cast<const v8bf*>(rp + 16);
  return u.v;
}

// Same A fragment but built from an f32 row-major tile with inline cvt (residual stays exact f32)
__device__ __forceinline__ v16bf build_a_frag_f32(const float* base, int ld, int m0, int k0, int lane) {
  int m  = m0 + (lane & 15);
  int hi = lane >> 4;
  const float* rp = base + m * ld + k0 + hi * 8;
  V16U u;
  #pragma unroll
  for (int g = 0; g < 2; ++g) {
    float4 f0 = *reinterpret_cast<const float4*>(rp + g * 16);
    float4 f1 = *reinterpret_cast<const float4*>(rp + g * 16 + 4);
    u.s[g * 8 + 0] = f2bf(f0.x); u.s[g * 8 + 1] = f2bf(f0.y);
    u.s[g * 8 + 2] = f2bf(f0.z); u.s[g * 8 + 3] = f2bf(f0.w);
    u.s[g * 8 + 4] = f2bf(f1.x); u.s[g * 8 + 5] = f2bf(f1.y);
    u.s[g * 8 + 6] = f2bf(f1.z); u.s[g * 8 + 7] = f2bf(f1.w);
  }
  return u.v;
}

// B fragment (32x16 bf16, K x N) from row-major "codes" E[n][k]: lane l -> col n0+(l&15);
// element e -> k = k0 + (e>>3)*16 + (l>>4)*8 + (e&7)  (two 16B loads, no transpose needed)
__device__ __forceinline__ v16bf load_b_frag(const unsigned short* base, int ld, int n0, int k0, int lane) {
  int n  = n0 + (lane & 15);
  int hi = lane >> 4;
  const unsigned short* rp = base + n * ld + k0 + hi * 8;
  V16 u;
  u.h[0] = *reinterpret_cast<const v8bf*>(rp);
  u.h[1] = *reinterpret_cast<const v8bf*>(rp + 16);
  return u.v;
}

__device__ __forceinline__ v8f wmma_bf16(v16bf a, v16bf b, v8f c) {
  return __builtin_amdgcn_wmma_f32_16x16x32_bf16(false, a, false, b, (short)0, c, false, false);
}

// ---------------- prep: bf16 copies of weights ----------------
__global__ void k_prep_w(const float* __restrict__ w_in, const float* __restrict__ w_out,
                         unsigned short* __restrict__ w_in_bf, unsigned short* __restrict__ w_out_bf) {
  int i = blockIdx.x * blockDim.x + threadIdx.x;   // 0 .. 131071
  w_in_bf[i]  = f2bf(w_in[i]);
  w_out_bf[i] = f2bf(w_out[i]);
}

// ---------------- prep: bf16 codebooks + ||e||^2 ----------------
__global__ void k_prep_emb(const float* __restrict__ emb, unsigned short* __restrict__ emb_bf,
                           float* __restrict__ e_sq) {
  __shared__ float red[256];
  int b = blockIdx.x;          // code row (q*K + k)
  int t = threadIdx.x;         // channel d
  float f = emb[(size_t)b * DCB + t];
  emb_bf[(size_t)b * DCB + t] = f2bf(f);
  red[t] = f * f;
  __syncthreads();
  for (int s = 128; s > 0; s >>= 1) {
    if (t < s) red[t] += red[t + s];
    __syncthreads();
  }
  if (t == 0) e_sq[b] = red[0];
}

// ---------------- input projection: R0 = X @ Win^T ----------------
__global__ void __launch_bounds__(256) k_proj_in(const float* __restrict__ x,
                                                 const unsigned short* __restrict__ w_in_bf,
                                                 float* __restrict__ R0) {
  extern __shared__ char smem[];
  unsigned short* Xbf = (unsigned short*)smem;     // 128 x 512 bf16 = 128KB
  int tid = threadIdx.x, wg = blockIdx.x;
  const float* xs = x + (size_t)wg * 128 * DIN;
  for (int i = 0; i < 256; ++i) {
    int idx = i * 256 + tid;
    Xbf[idx] = f2bf(xs[idx]);
  }
  __syncthreads();
  int lane = tid & 31, wave = tid >> 5;
  int m0 = wave * 16, hi = lane >> 4, l15 = lane & 15;
  for (int nt = 0; nt < 16; ++nt) {                // N = 256
    v8f acc = {};
    #pragma unroll
    for (int kk = 0; kk < 16; ++kk) {              // K = 512
      v16bf a = load_a_frag(Xbf, DIN, m0, kk * 32, lane);
      v16bf b = load_b_frag(w_in_bf, DIN, nt * 16, kk * 32, lane);
      acc = wmma_bf16(a, b, acc);
    }
    #pragma unroll
    for (int v = 0; v < 8; ++v) {
      int row = wg * 128 + m0 + hi * 8 + v;
      R0[(size_t)row * DCB + nt * 16 + l15] = acc[v];
    }
  }
}

// ---------------- fused 8-round residual VQ ----------------
// LDS: residF 128KB @0, Bs 16KB @131072, esqS 128B @147456, idxs 512B @147584 -> 148096 B
// <=160KB  =>  two workgroups co-resident per WGP (4 waves/SIMD32) so one WG's staging
// overlaps the other WG's WMMA phase.
#define LDS_BS_OFF   131072
#define LDS_ESQ_OFF  147456
#define LDS_IDX_OFF  147584
#define LDS_RVQ_SIZE 148096

__global__ void __launch_bounds__(256) k_rvq(const float* __restrict__ R0,
                                             const unsigned short* __restrict__ emb_bf,
                                             const float* __restrict__ e_sq,
                                             const float* __restrict__ emb_f32,
                                             float* __restrict__ Rfin) {
  extern __shared__ char smem[];
  float*          residF = (float*)smem;                          // 128*256 f32
  unsigned short* Bs     = (unsigned short*)(smem + LDS_BS_OFF);  //  32*256 bf16
  float*          esqS   = (float*)(smem + LDS_ESQ_OFF);          //  32 f32
  int*            idxs   = (int*)(smem + LDS_IDX_OFF);            // 128 int

  int tid = threadIdx.x, wg = blockIdx.x;
  int lane = tid & 31, wave = tid >> 5;
  int m0 = wave * 16, hi = lane >> 4, l15 = lane & 15;
  const float* r0 = R0 + (size_t)wg * 128 * DCB;

  for (int i = 0; i < 128; ++i) {
    int idx = i * 256 + tid;
    residF[idx] = r0[idx];
  }
  __syncthreads();

  for (int q = 0; q < NQ; ++q) {
    const unsigned short* ebq = emb_bf + (size_t)q * KC * DCB;
    const float*        esq_q = e_sq + q * KC;

    // rebuild this wave's A fragments (16 rows x 256 k) from the exact f32 residual
    V16 a[8];
    #pragma unroll
    for (int kk = 0; kk < 8; ++kk) a[kk].v = build_a_frag_f32(residF, DCB, m0, kk * 32, lane);

    float bd[8]; int bi[8];
    #pragma unroll
    for (int v = 0; v < 8; ++v) { bd[v] = 3.4e38f; bi[v] = 0; }

    for (int chunk = 0; chunk < 64; ++chunk) {     // 64 chunks x 32 codes = 2048
      __syncthreads();
#if __has_builtin(__builtin_amdgcn_tensor_load_to_lds)
      // Tensor Data Mover: DMA the 16KB codebook chunk into LDS (1-D tile, 8B elems),
      // issued once by wave 0; TDM ignores EXEC. Frees VALU/VMEM pipes for the WMMA WG.
      if (wave == 0) {
        unsigned long long ga =
            (unsigned long long)(uintptr_t)(ebq + (size_t)chunk * 32 * DCB);
        u32x4 g0 = { 1u,                                   // count=1, user descriptor
                     (unsigned)LDS_BS_OFF,                 // lds_addr (bytes)
                     (unsigned)(ga & 0xFFFFFFFFu),         // global_addr[31:0]
                     (unsigned)((ga >> 32) & 0x01FFFFFFu) | 0x80000000u }; // addr[56:32] | type=2
        i32x8 g1 = { (int)(3u << 16),                      // data_size = 8B qwords
                     (int)(2048u << 16),                   // tensor_dim0[15:0] in bits63:48
                     0,                                    // tensor_dim0 hi / tensor_dim1 lo
                     (int)(2048u << 16),                   // tile_dim0 = 2048 qwords (16KB)
                     0,                                    // tile_dim1/2 = 0 (1-D)
                     2048,                                 // tensor_dim0_stride lo
                     0, 0 };
        i32x4 gz = { 0, 0, 0, 0 };
#if defined(__clang_major__) && __clang_major__ >= 23
        i32x8 gz8 = { 0, 0, 0, 0, 0, 0, 0, 0 };
        __builtin_amdgcn_tensor_load_to_lds(g0, g1, gz, gz, gz8, 0);
#else
        __builtin_amdgcn_tensor_load_to_lds(g0, g1, gz, gz, 0);
#endif
        __builtin_amdgcn_s_wait_tensorcnt(0);
      }
      if (tid < 32) esqS[tid] = esq_q[chunk * 32 + tid];
#else
      {
        const float4* src = (const float4*)(ebq + (size_t)chunk * 32 * DCB);
        float4*       dst = (float4*)Bs;
        #pragma unroll
        for (int i = 0; i < 4; ++i) dst[i * 256 + tid] = src[i * 256 + tid];
        if (tid < 32) esqS[tid] = esq_q[chunk * 32 + tid];
        if (chunk + 1 < 64)
          __builtin_prefetch(ebq + (size_t)(chunk + 1) * 32 * DCB + tid * 32, 0, 3);
      }
#endif
      __syncthreads();
      #pragma unroll
      for (int ns = 0; ns < 2; ++ns) {             // 2 subtiles of 16 codes
        v8f acc = {};
        #pragma unroll
        for (int kk = 0; kk < 8; ++kk)             // K(dim) = 256
          acc = wmma_bf16(a[kk].v, load_b_frag(Bs, DCB, ns * 16, kk * 32, lane), acc);
        float es = esqS[ns * 16 + l15];
        int   ng = chunk * 32 + ns * 16 + l15;
        #pragma unroll
        for (int v = 0; v < 8; ++v) {
          float d = fmaf(-2.0f, acc[v], es);       // ||x||^2 constant per row: dropped
          if (d < bd[v]) { bd[v] = d; bi[v] = ng; }
        }
      }
    }
    // cross-lane argmin over the 16 lanes holding the 16 column candidates
    #pragma unroll
    for (int v = 0; v < 8; ++v) {
      float d = bd[v]; int ix = bi[v];
      #pragma unroll
      for (int m = 1; m < 16; m <<= 1) {
        float od = __shfl_xor(d, m, 32);
        int   oi = __shfl_xor(ix, m, 32);
        if (od < d || (od == d && oi < ix)) { d = od; ix = oi; }
      }
      if (l15 == 0) idxs[m0 + hi * 8 + v] = ix;
    }
    __syncthreads();
    // gather exact fp32 code rows, update f32 residual
    const float* ef = emb_f32 + (size_t)q * KC * DCB;
    for (int r = 0; r < 128; ++r) {
      int code = idxs[r];
      float e  = ef[(size_t)code * DCB + tid];
      residF[r * DCB + tid] -= e;
    }
    __syncthreads();
  }

  float* rf = Rfin + (size_t)wg * 128 * DCB;
  for (int i = 0; i < 128; ++i) {
    int idx = i * 256 + tid;
    rf[idx] = residF[idx];
  }
}

// ---------------- output projection: out = (R0 - Rfin) @ Wout^T ----------------
__global__ void __launch_bounds__(256) k_proj_out(const float* __restrict__ R0,
                                                  const float* __restrict__ Rfin,
                                                  const unsigned short* __restrict__ w_out_bf,
                                                  float* __restrict__ out) {
  extern __shared__ char smem[];
  unsigned short* Abf = (unsigned short*)smem;     // 128 x 256 bf16 = 64KB
  int tid = threadIdx.x, wg = blockIdx.x;
  const float* a0 = R0   + (size_t)wg * 128 * DCB;
  const float* a1 = Rfin + (size_t)wg * 128 * DCB;
  for (int i = 0; i < 128; ++i) {
    int idx = i * 256 + tid;
    Abf[idx] = f2bf(a0[idx] - a1[idx]);            // sum of quantized = R0 - Rfin
  }
  __syncthreads();
  int lane = tid & 31, wave = tid >> 5;
  int m0 = wave * 16, hi = lane >> 4, l15 = lane & 15;
  V16 a[8];
  #pragma unroll
  for (int kk = 0; kk < 8; ++kk) a[kk].v = load_a_frag(Abf, DCB, m0, kk * 32, lane);
  for (int nt = 0; nt < 32; ++nt) {                // N = 512
    v8f acc = {};
    #pragma unroll
    for (int kk = 0; kk < 8; ++kk)                 // K = 256
      acc = wmma_bf16(a[kk].v, load_b_frag(w_out_bf, DCB, nt * 16, kk * 32, lane), acc);
    #pragma unroll
    for (int v = 0; v < 8; ++v) {
      int row = wg * 128 + m0 + hi * 8 + v;
      out[(size_t)row * DIN + nt * 16 + l15] = acc[v];
    }
  }
}

extern "C" void kernel_launch(void* const* d_in, const int* in_sizes, int n_in,
                              void* d_out, int out_size, void* d_ws, size_t ws_size,
                              hipStream_t stream) {
  (void)in_sizes; (void)n_in; (void)out_size; (void)ws_size;
  const float* x     = (const float*)d_in[0];
  const float* w_in  = (const float*)d_in[1];
  const float* w_out = (const float*)d_in[2];
  const float* emb   = (const float*)d_in[3];

  char* ws = (char*)d_ws;
  float* R0   = (float*)ws;                       ws += (size_t)TT * DCB * 4;        // 32 MB
  float* Rfin = (float*)ws;                       ws += (size_t)TT * DCB * 4;        // 32 MB
  unsigned short* emb_bf = (unsigned short*)ws;   ws += (size_t)NQ * KC * DCB * 2;   //  8 MB
  float* e_sq = (float*)ws;                       ws += (size_t)NQ * KC * 4;         // 64 KB
  unsigned short* w_in_bf  = (unsigned short*)ws; ws += (size_t)DCB * DIN * 2;       // 256 KB
  unsigned short* w_out_bf = (unsigned short*)ws; ws += (size_t)DIN * DCB * 2;       // 256 KB

  (void)hipFuncSetAttribute((const void*)k_proj_in,  hipFuncAttributeMaxDynamicSharedMemorySize, 131072);
  (void)hipFuncSetAttribute((const void*)k_rvq,      hipFuncAttributeMaxDynamicSharedMemorySize, LDS_RVQ_SIZE);
  (void)hipFuncSetAttribute((const void*)k_proj_out, hipFuncAttributeMaxDynamicSharedMemorySize, 65536);

  k_prep_w  <<<(DCB * DIN) / 256, 256, 0, stream>>>(w_in, w_out, w_in_bf, w_out_bf);
  k_prep_emb<<<NQ * KC,           256, 0, stream>>>(emb, emb_bf, e_sq);
  k_proj_in <<<TT / 128, 256, 131072, stream>>>(x, w_in_bf, R0);
  k_rvq     <<<TT / 128, 256, LDS_RVQ_SIZE, stream>>>(R0, emb_bf, e_sq, emb, Rfin);
  k_proj_out<<<TT / 128, 256, 65536,  stream>>>(R0, Rfin, w_out_bf, (float*)d_out);
}
